// StandardMultiheadAttention_71957882077244
// MI455X (gfx1250) — matrix operand order repaired and checked by reading, not compiled, not run on hardware
//
#include <hip/hip_runtime.h>

typedef __bf16 bf16;
typedef __attribute__((ext_vector_type(16))) __bf16 v16bf;
typedef __attribute__((ext_vector_type(8)))  __bf16 v8bf;
typedef __attribute__((ext_vector_type(8)))  float  v8f;
typedef __attribute__((ext_vector_type(4)))  int    v4i;

#define BATCH 2
#define SEQ   4096
#define CH    512
#define NH    8
#define HD    64
#define NQK   1536         // 3*CH
#define SCALEF 0.125f      // 64^-0.5

// LDS row strides padded to non-power-of-two multiples of 16B (bank spread)
#define KLD 72             // K tile row stride (bf16 elems), 144 B
#define VLD 40             // V^T tile row stride, 80 B
#define PLD 40             // P tile row stride, 80 B

#if defined(__has_builtin)
#if __has_builtin(__builtin_amdgcn_global_load_async_to_lds_b128) && \
    __has_builtin(__builtin_amdgcn_s_wait_asynccnt)
#define HAVE_ASYNC 1
#endif
#endif
#ifndef HAVE_ASYNC
#define HAVE_ASYNC 0
#endif

typedef v4i __attribute__((address_space(1)))* gptr_v4i;   // global
typedef v4i __attribute__((address_space(3)))* lptr_v4i;   // LDS

// ---------------------------------------------------------------------------
// Fragment helpers for V_WMMA_F32_16X16X32_BF16 (wave32).
// A (16x32, MxK): lane row = lane&15; lanes<16 hold K {k0..k0+7, k0+16..k0+23},
//                 lanes>=16 hold K {k0+8..k0+15, k0+24..k0+31}.
// B (32x16, KxN): lane col = lane&15; same K split as A (lanes<16 hold
//                 K {k0..k0+7, k0+16..k0+23}, lanes>=16 the rest).  B kept transposed (N-major)
//                 so each lane reads two runs of 8 contiguous bf16.
// C/D (16x16):    VGPR r, lanes<16 -> (M=r, N=lane); lanes>=16 -> (M=r+8).
// ---------------------------------------------------------------------------
static __device__ __forceinline__ v16bf concat8(v8bf a0, v8bf a1) {
  v16bf a;
#pragma unroll
  for (int i = 0; i < 8; ++i) { a[i] = a0[i]; a[8 + i] = a1[i]; }
  return a;
}

static __device__ __forceinline__ v16bf load_a(const bf16* base, int ld,
                                               int row0, int k0, int lane) {
  const int row = row0 + (lane & 15);
  const int kb  = k0 + ((lane & 16) ? 8 : 0);
  const bf16* p = base + row * ld + kb;
  v8bf a0 = *(const v8bf*)(p);
  v8bf a1 = *(const v8bf*)(p + 16);
  return concat8(a0, a1);
}

// bt is the (N-major) transposed B: bt[n][k], leading dim ld (= K extent)
static __device__ __forceinline__ v16bf load_bt(const bf16* bt, int ld,
                                                int n0, int k0, int lane) {
  const int n  = n0 + (lane & 15);
  const int kb = k0 + ((lane & 16) ? 8 : 0);
  const bf16* p = bt + n * ld + kb;
  v8bf b0 = *(const v8bf*)(p);
  v8bf b1 = *(const v8bf*)(p + 16);
  return concat8(b0, b1);
}

static __device__ __forceinline__ v8f wmma_bf16(v16bf a, v16bf b, v8f c) {
  return __builtin_amdgcn_wmma_f32_16x16x32_bf16(false, a, false, b,
                                                 (short)0, c, false, false);
}

// 16-byte async (or fallback synchronous) global -> LDS copy
static __device__ __forceinline__ void cp16_g2l(bf16* lds_dst,
                                                const bf16* g_src) {
#if HAVE_ASYNC
  __builtin_amdgcn_global_load_async_to_lds_b128(
      (gptr_v4i)(unsigned long long)(g_src),
      (lptr_v4i)(unsigned)(unsigned long long)(lds_dst),
      0, 0);
#else
  *(v8bf*)lds_dst = *(const v8bf*)g_src;
#endif
}

static __device__ __forceinline__ void async_wait0() {
#if HAVE_ASYNC
  __builtin_amdgcn_s_wait_asynccnt(0);
#endif
}

// ---------------------------------------------------------------------------
// Prep kernels
// ---------------------------------------------------------------------------
__global__ void cvt_f32_bf16(const float* __restrict__ in,
                             bf16* __restrict__ out, int n) {
  int i = blockIdx.x * blockDim.x + threadIdx.x;
  if (i < n) out[i] = (bf16)in[i];
}

// w[k][n] (K rows, N cols, row-major) -> wt[n][k] bf16
__global__ void transpose_to_bf16(const float* __restrict__ w,
                                  bf16* __restrict__ wt, int K, int N) {
  int i = blockIdx.x * blockDim.x + threadIdx.x;
  if (i < K * N) {
    int k = i / N, n = i % N;
    wt[n * K + k] = (bf16)w[i];
  }
}

// ---------------------------------------------------------------------------
// QKV projection: xb[8192,512] @ wqt^T -> Q,K [B,H,L,D] bf16; V^T [B,H,D,L].
// Each wave computes a 16x64 strip: one A fragment feeds 4 accumulators.
// ---------------------------------------------------------------------------
__global__ void __launch_bounds__(256) qkv_gemm(const bf16* __restrict__ xb,
                                                const bf16* __restrict__ wqt,
                                                bf16* __restrict__ Qb,
                                                bf16* __restrict__ Kb,
                                                bf16* __restrict__ Vtb) {
  const int lane = threadIdx.x & 31;
  const int tile = blockIdx.x * 8 + (threadIdx.x >> 5);
  const int NT = NQK / 64;          // 24 strips of 64 columns
  const int ti = tile / NT, tj = tile % NT;

  v8f cc[4] = {{}, {}, {}, {}};
#pragma unroll 2
  for (int kk = 0; kk < CH; kk += 32) {
    v16bf a = load_a(xb, CH, ti * 16, kk, lane);
#pragma unroll
    for (int s = 0; s < 4; ++s)
      cc[s] = wmma_bf16(a, load_bt(wqt, CH, tj * 64 + s * 16, kk, lane), cc[s]);
  }

#pragma unroll
  for (int s = 0; s < 4; ++s) {
    const int n     = tj * 64 + s * 16 + (lane & 15);  // fixed per lane
    const int which = n >> 9;                          // 0:Q 1:K 2:V
    const int h     = (n & 511) >> 6;
    const int d     = n & 63;
#pragma unroll
    for (int r = 0; r < 8; ++r) {
      const int m  = ti * 16 + r + ((lane & 16) ? 8 : 0);
      const int b_ = m >> 12;
      const int l  = m & (SEQ - 1);
      bf16 v = (bf16)cc[s][r];
      if (which == 0)      Qb [((b_ * NH + h) * SEQ + l) * HD + d] = v;
      else if (which == 1) Kb [((b_ * NH + h) * SEQ + l) * HD + d] = v;
      else                 Vtb[((b_ * NH + h) * HD + d) * SEQ + l] = v;
    }
  }
}

// ---------------------------------------------------------------------------
// Flash attention. Block = 8 waves x 16 q-rows = 128 rows. K/V tiles for the
// whole block are staged into LDS with double-buffered async global->LDS
// copies (ASYNCcnt); each wave runs online softmax on its 16 rows. LDS rows
// padded (non-power-of-two strides) to avoid bank conflicts.
// ---------------------------------------------------------------------------
__global__ void __launch_bounds__(256) flash_attn(const bf16* __restrict__ Qb,
                                                  const bf16* __restrict__ Kb,
                                                  const bf16* __restrict__ Vtb,
                                                  float* __restrict__ Ob) {
  __shared__ bf16 kbuf[2][32 * KLD];  // 2 x 4.5 KB : K rows j..j+31, row-major
  __shared__ bf16 vbuf[2][HD * VLD];  // 2 x 5 KB   : V^T rows d, cols j..j+31
  __shared__ bf16 pbuf[8][16 * PLD];  // 10 KB      : per-wave P tile

  const int tid  = threadIdx.x;
  const int lane = tid & 31;
  const int wave = tid >> 5;
  const int bh   = blockIdx.y;

  const bf16* q  = Qb  + (size_t)bh * SEQ * HD;
  const bf16* kg = Kb  + (size_t)bh * SEQ * HD;
  const bf16* vg = Vtb + (size_t)bh * HD * SEQ;
  float*      o  = Ob  + (size_t)bh * SEQ * HD;

  const int q0 = blockIdx.x * 128 + wave * 16;
  const v16bf qa0 = load_a(q, HD, q0, 0, lane);
  const v16bf qa1 = load_a(q, HD, q0, 32, lane);

  v8f oacc0 = {}, oacc1 = {}, oacc2 = {}, oacc3 = {};
  float rmax[8], rsum[8];
#pragma unroll
  for (int r = 0; r < 8; ++r) { rmax[r] = -3.0e38f; rsum[r] = 0.f; }

  bf16* pl = pbuf[wave];

  // stage(buf, j): 256 threads x 16B cover the 4KB K tile and 4KB V tile
  auto stage = [&](int buf, int j) {
    const int kr = tid >> 3, kc = (tid & 7) * 8;     // 32 rows x 8 chunks
    cp16_g2l(&kbuf[buf][kr * KLD + kc], kg + (size_t)(j + kr) * HD + kc);
    const int d = tid >> 2, c4 = (tid & 3) * 8;      // 64 rows x 4 chunks
    cp16_g2l(&vbuf[buf][d * VLD + c4], vg + (size_t)d * SEQ + j + c4);
  };

  stage(0, 0);
  int cur = 0;
  for (int j = 0; j < SEQ; j += 32, cur ^= 1) {
    async_wait0();
    __syncthreads();                       // staged tiles visible; prev reads done
    if (j + 32 < SEQ) stage(cur ^ 1, j + 32);

    const bf16* kl = kbuf[cur];
    const bf16* vl = vbuf[cur];

    // ---- S = Q K^T for 16x32 key block (K-dim 64 = 2x32), from LDS ----
    v8f s0 = {}, s1 = {};
    s0 = wmma_bf16(qa0, load_bt(kl, KLD, 0,   0, lane), s0);
    s0 = wmma_bf16(qa1, load_bt(kl, KLD, 0,  32, lane), s0);
    s1 = wmma_bf16(qa0, load_bt(kl, KLD, 16,  0, lane), s1);
    s1 = wmma_bf16(qa1, load_bt(kl, KLD, 16, 32, lane), s1);

    // ---- online softmax (rows live in 16-lane halves of C-layout) ----
#pragma unroll
    for (int r = 0; r < 8; ++r) {
      float a0 = s0[r] * SCALEF, a1 = s1[r] * SCALEF;
      float mx = fmaxf(a0, a1);
#pragma unroll
      for (int off = 1; off < 16; off <<= 1)
        mx = fmaxf(mx, __shfl_xor(mx, off, 32));
      const float nm = fmaxf(rmax[r], mx);
      const float sc = __expf(rmax[r] - nm);
      rmax[r] = nm;
      const float p0 = __expf(a0 - nm);
      const float p1 = __expf(a1 - nm);
      float ps = p0 + p1;
#pragma unroll
      for (int off = 1; off < 16; off <<= 1)
        ps += __shfl_xor(ps, off, 32);
      rsum[r] = rsum[r] * sc + ps;
      oacc0[r] *= sc; oacc1[r] *= sc; oacc2[r] *= sc; oacc3[r] *= sc;
      // C-layout -> row-major P tile in per-wave LDS
      const int m = r + ((lane & 16) ? 8 : 0);
      pl[m * PLD + (lane & 15)]      = (bf16)p0;
      pl[m * PLD + 16 + (lane & 15)] = (bf16)p1;
    }

    // ---- O += P V  (A from LDS, B from staged V^T, 4 N-tiles) ----
    v16bf pa = load_a(pl, PLD, 0, 0, lane);  // DS ops in-order per wave
    oacc0 = wmma_bf16(pa, load_bt(vl, VLD,  0, 0, lane), oacc0);
    oacc1 = wmma_bf16(pa, load_bt(vl, VLD, 16, 0, lane), oacc1);
    oacc2 = wmma_bf16(pa, load_bt(vl, VLD, 32, 0, lane), oacc2);
    oacc3 = wmma_bf16(pa, load_bt(vl, VLD, 48, 0, lane), oacc3);
  }

  // ---- normalize and store O [B,H,L,D] fp32 ----
  const int dcol = lane & 15;
#pragma unroll
  for (int r = 0; r < 8; ++r) {
    const int m = q0 + r + ((lane & 16) ? 8 : 0);
    const float inv = 1.0f / rsum[r];
    o[m * HD +  0 + dcol] = oacc0[r] * inv;
    o[m * HD + 16 + dcol] = oacc1[r] * inv;
    o[m * HD + 32 + dcol] = oacc2[r] * inv;
    o[m * HD + 48 + dcol] = oacc3[r] * inv;
  }
}

// ---------------------------------------------------------------------------
// Output projection: gather head-interleaved fp32 O -> bf16 A frags in
// register, times w_out^T, + bias, fp32 out [B,L,C]. 16x64 strip per wave.
// ---------------------------------------------------------------------------
static __device__ __forceinline__ v8bf cvt8(const float* __restrict__ p) {
  v8bf r;
#pragma unroll
  for (int i = 0; i < 8; ++i) r[i] = (bf16)p[i];
  return r;
}

__global__ void __launch_bounds__(256) out_gemm(const float* __restrict__ Ob,
                                                const bf16* __restrict__ wot,
                                                const float* __restrict__ bias,
                                                float* __restrict__ out) {
  const int lane = threadIdx.x & 31;
  const int tile = blockIdx.x * 8 + (threadIdx.x >> 5);
  const int NT = CH / 64;                 // 8 strips
  const int ti = tile / NT, tj = tile % NT;

  const int row = ti * 16 + (lane & 15);
  const int b_  = row >> 12;
  const int l   = row & (SEQ - 1);
  const float* abase = Ob + ((size_t)b_ * NH) * SEQ * HD + (size_t)l * HD;

  v8f cc[4] = {{}, {}, {}, {}};
  for (int kk = 0; kk < CH; kk += 32) {
    const int kb = kk + ((lane & 16) ? 8 : 0);
    const int k0 = kb, k1 = kb + 16;      // 8-aligned -> stays inside a head
    const float* p0 = abase + (size_t)(k0 >> 6) * SEQ * HD + (k0 & 63);
    const float* p1 = abase + (size_t)(k1 >> 6) * SEQ * HD + (k1 & 63);
    v16bf a = concat8(cvt8(p0), cvt8(p1));
#pragma unroll
    for (int s = 0; s < 4; ++s)
      cc[s] = wmma_bf16(a, load_bt(wot, CH, tj * 64 + s * 16, kk, lane), cc[s]);
  }

#pragma unroll
  for (int s = 0; s < 4; ++s) {
    const int n  = tj * 64 + s * 16 + (lane & 15);
    const float bv = bias[n];
#pragma unroll
    for (int r = 0; r < 8; ++r) {
      const int m = ti * 16 + r + ((lane & 16) ? 8 : 0);
      out[(size_t)m * CH + n] = cc[s][r] + bv;
    }
  }
}

// ---------------------------------------------------------------------------
extern "C" void kernel_launch(void* const* d_in, const int* in_sizes, int n_in,
                              void* d_out, int out_size, void* d_ws,
                              size_t ws_size, hipStream_t stream) {
  (void)in_sizes; (void)n_in; (void)out_size; (void)ws_size;
  const float* x    = (const float*)d_in[0];
  const float* wqkv = (const float*)d_in[1];   // [512, 1536] (in x out)
  const float* wout = (const float*)d_in[2];   // [512, 512]
  const float* bout = (const float*)d_in[3];   // [512]
  float* out = (float*)d_out;

  const size_t M = (size_t)BATCH * SEQ;        // 8192
  char* ws = (char*)d_ws;
  size_t off = 0;
  auto alloc = [&](size_t bytes) -> char* {
    char* p = ws + off;
    off += (bytes + 255) & ~(size_t)255;
    return p;
  };
  bf16*  xb  = (bf16*)alloc(M * CH * sizeof(bf16));            // 8 MB
  bf16*  wqt = (bf16*)alloc((size_t)NQK * CH * sizeof(bf16));  // 1.5 MB
  bf16*  wot = (bf16*)alloc((size_t)CH * CH * sizeof(bf16));   // 0.5 MB
  bf16*  Qb  = (bf16*)alloc(M * CH * sizeof(bf16));            // 8 MB
  bf16*  Kb  = (bf16*)alloc(M * CH * sizeof(bf16));            // 8 MB
  bf16*  Vtb = (bf16*)alloc(M * CH * sizeof(bf16));            // 8 MB
  float* Ob  = (float*)alloc(M * CH * sizeof(float));          // 16 MB

  const int nX = (int)(M * CH);
  cvt_f32_bf16<<<(nX + 255) / 256, 256, 0, stream>>>(x, xb, nX);
  transpose_to_bf16<<<(CH * NQK + 255) / 256, 256, 0, stream>>>(wqkv, wqt, CH, NQK);
  transpose_to_bf16<<<(CH * CH + 255) / 256, 256, 0, stream>>>(wout, wot, CH, CH);

  qkv_gemm<<<(int)((M / 16) * (NQK / 64)) / 8, 256, 0, stream>>>(xb, wqt, Qb, Kb, Vtb);

  dim3 ag(SEQ / 128, BATCH * NH);
  flash_attn<<<ag, 256, 0, stream>>>(Qb, Kb, Vtb, Ob);

  out_gemm<<<(int)((M / 16) * (CH / 64)) / 8, 256, 0, stream>>>(Ob, wot, bout, out);
}
